// gatcase30_80814104641731
// MI455X (gfx1250) — matrix operand order repaired and checked by reading, not compiled, not run on hardware
//
#include <hip/hip_runtime.h>
#include <cstdint>

typedef __attribute__((ext_vector_type(16))) _Float16 v16h;
typedef __attribute__((ext_vector_type(8)))  float    v8f;

#define HID 256
#define CAP 96
#define BM 128
#define BN 128
#define BK 32
#define LDA 32
#define LDB 34   // 68B LDS row stride -> bank-conflict free (matches TDM pad)

#if __has_builtin(__builtin_amdgcn_tensor_load_to_lds)
#define HAVE_TDM 1
#endif

typedef __attribute__((ext_vector_type(4))) unsigned int tdm_u4;
typedef __attribute__((ext_vector_type(8))) int          tdm_i8;
typedef __attribute__((ext_vector_type(4))) int          tdm_i4;
typedef __attribute__((address_space(3))) _Float16       lds_f16;

// ---------------------------------------------------------------------------
// Tiled WMMA GEMM: C[M,N] = act( A[M,K] @ Wt^T + bias ), Wt is f16 [N][K]
// (pre-converted, transposed weights). A fp32 -> f16 staged through LDS.
// fp32 accumulation via v_wmma_f32_16x16x32_f16; 8 waves, wave tile 64x32.
// Double-buffered LDS: B tiles DMA'd by the Tensor Data Mover (with LDS
// padding pad_interval=16DW/pad_amount=1DW -> 68B row stride == LDB) and A
// tiles loaded to regs for tile k+1 while WMMAs run on tile k.
// GATHER: A row r, col c reads A[gb{0,1}[r] + (c & 255)] (edge-pair concat).
// ---------------------------------------------------------------------------
template <bool GATHER>
__global__ __launch_bounds__(256) void wmma_gemm_t(
    const float* __restrict__ A, const _Float16* __restrict__ Wt,
    const float* __restrict__ bias, float* __restrict__ C,
    int M, int N, int K, int act,
    const int* __restrict__ gb0, const int* __restrict__ gb1)
{
  __shared__ _Float16 As[2 * BM * LDA];
  __shared__ _Float16 Bs[2 * BN * LDB];
  __shared__ int rb0[BM];
  __shared__ int rb1[BM];

  const int t    = threadIdx.x;
  const int lane = t & 31;
  const int wave = t >> 5;
  const int wm   = wave >> 2;   // 0..1 : row half
  const int wn   = wave & 3;    // 0..3 : col quarter
  const int m0   = blockIdx.x * BM;
  const int n0   = blockIdx.y * BN;

  if (GATHER) {
    if (t < BM) { rb0[t] = gb0[m0 + t]; rb1[t] = gb1[m0 + t]; }
  }
  __syncthreads();   // rb visible before first A issue

  v8f acc[4][2];
#pragma unroll
  for (int i = 0; i < 4; ++i)
#pragma unroll
    for (int j = 0; j < 2; ++j)
#pragma unroll
      for (int q = 0; q < 8; ++q) acc[i][j][q] = 0.0f;

  const int fm = lane & 31 & 15;
  const int hi = lane >> 4;

#ifdef HAVE_TDM
  const unsigned bsOff0 = (unsigned)(size_t)(lds_f16*)&Bs[0];
  const unsigned bsOff1 = (unsigned)(size_t)(lds_f16*)&Bs[BN * LDB];
#endif

  float4 va[4];

  // ---- issue B tile DMA for k0 into buffer buf ----
  auto issueB = [&](int k0, int buf) {
#ifdef HAVE_TDM
    if (wave == 0) {
      uint64_t ga = (uint64_t)(size_t)(Wt + ((size_t)n0 * K + k0));
      tdm_u4 g0;
      g0.x = 1u;                                   // count=1, normal mode
      g0.y = buf ? bsOff1 : bsOff0;                // lds_addr
      g0.z = (unsigned)ga;                         // global_addr[31:0]
      g0.w = (unsigned)(ga >> 32) | (2u << 30);    // global_addr[56:32] | type=2
      tdm_i8 g1 = {0, 0, 0, 0, 0, 0, 0, 0};
      g1[0] = (1 << 16)      // data_size = 2 bytes
            | (1 << 20)      // pad_enable
            | (3 << 22)      // pad_interval: every 16 DWORDs (=64B row)
            | (0 << 25);     // pad_amount: 1 DWORD -> 68B LDS row stride
      g1[1] = (int)(32u << 16);    // tensor_dim0 = 32
      g1[2] = (int)(128u << 16);   // tensor_dim1 = 128
      g1[3] = (int)(32u << 16);    // tile_dim0 = 32 elements (64B)
      g1[4] = 128;                 // tile_dim1 = 128 rows
      g1[5] = K;                   // tensor_dim0_stride (elements)
      tdm_i4 gz = {0, 0, 0, 0};
#if defined(__clang_major__) && __clang_major__ >= 23
      tdm_i8 gz8 = {0, 0, 0, 0, 0, 0, 0, 0};
      __builtin_amdgcn_tensor_load_to_lds(g0, g1, gz, gz, gz8, 0);
#else
      __builtin_amdgcn_tensor_load_to_lds(g0, g1, gz, gz, 0);
#endif
    }
#else
    // fallback: per-thread f16 copy of the B tile
    _Float16* bb = &Bs[buf * BN * LDB];
    for (int i = 0; i < 2; ++i) {
      int flat = (t + 256 * i) * 8;
      int nn = flat >> 5, kk = flat & 31;
      const _Float16* sp = Wt + (size_t)(n0 + nn) * K + k0 + kk;
      _Float16* dp = bb + nn * LDB + kk;
#pragma unroll
      for (int q = 0; q < 8; ++q) dp[q] = sp[q];
    }
#endif
  };

  // ---- phase 1: issue 4 independent global A loads into regs ----
  auto issueA = [&](int k0) {
#pragma unroll
    for (int i = 0; i < 4; ++i) {
      int flat = (t + 256 * i) * 4;
      int r  = flat >> 5;
      int kk = flat & 31;
      if (!GATHER) {
        va[i] = *(const float4*)(A + (size_t)(m0 + r) * K + (k0 + kk));
      } else {
        int c = k0 + kk;
        int base = (c < HID) ? rb0[r] : rb1[r];
        va[i] = *(const float4*)(A + base + (c & (HID - 1)));
      }
    }
  };

  // ---- phase 2: convert + store A regs into LDS buffer buf ----
  auto storeA = [&](int buf) {
    _Float16* ab = &As[buf * BM * LDA];
#pragma unroll
    for (int i = 0; i < 4; ++i) {
      int flat = (t + 256 * i) * 4;
      int r  = flat >> 5;
      int kk = flat & 31;
      _Float16* d = ab + r * LDA + kk;
      d[0] = (_Float16)va[i].x; d[1] = (_Float16)va[i].y;
      d[2] = (_Float16)va[i].z; d[3] = (_Float16)va[i].w;
    }
  };

  const int nTiles = K / BK;

  // ---- prologue: stage tile 0 into buffer 0 ----
  issueB(0, 0);
  issueA(0);
  storeA(0);
#ifdef HAVE_TDM
  __builtin_amdgcn_s_wait_tensorcnt(0);
#endif
  __syncthreads();

  int p = 0;
  for (int kt = 0; kt < nTiles; ++kt) {
    const bool more = (kt + 1) < nTiles;
    if (more) {
      issueB((kt + 1) * BK, p ^ 1);   // DMA next B tile (async)
      issueA((kt + 1) * BK);          // next A tile -> regs (async)
    }

    // ---- fragments from buffer p (ISA 16-bit A 16x32 / B 32x16 layouts) ----
    const _Float16* ab = &As[p * BM * LDA];
    const _Float16* bb = &Bs[p * BN * LDB];
    v16h af[4], bf[2];
#pragma unroll
    for (int i = 0; i < 4; ++i) {
      const _Float16* ap = ab + (wm * 64 + i * 16 + fm) * LDA;
#pragma unroll
      for (int q = 0; q < 8; ++q) {
        int kb = ((q >= 4) ? 16 : 0) + hi * 8 + (q & 3) * 2;
        af[i][2 * q]     = ap[kb];
        af[i][2 * q + 1] = ap[kb + 1];
      }
    }
#pragma unroll
    for (int j = 0; j < 2; ++j) {
      const _Float16* bp = bb + (wn * 32 + j * 16 + fm) * LDB;
#pragma unroll
      for (int q = 0; q < 8; ++q) {
        int kb = ((q >= 4) ? 16 : 0) + hi * 8 + (q & 3) * 2;
        bf[j][2 * q]     = bp[kb];
        bf[j][2 * q + 1] = bp[kb + 1];
      }
    }
#pragma unroll
    for (int i = 0; i < 4; ++i)
#pragma unroll
      for (int j = 0; j < 2; ++j)
        acc[i][j] = __builtin_amdgcn_wmma_f32_16x16x32_f16(
            false, af[i], false, bf[j], (short)0, acc[i][j], false, false);

    if (more) {
      storeA(p ^ 1);                  // cvt + LDS store behind the WMMAs
#ifdef HAVE_TDM
      __builtin_amdgcn_s_wait_tensorcnt(0);
#endif
    }
    __syncthreads();
    p ^= 1;
  }

  // ---- epilogue: bias + optional leaky-relu(0.01) ----
#pragma unroll
  for (int i = 0; i < 4; ++i) {
#pragma unroll
    for (int j = 0; j < 2; ++j) {
      int col = n0 + wn * 32 + j * 16 + fm;
      float bv = bias[col];
#pragma unroll
      for (int r = 0; r < 8; ++r) {
        int row = m0 + wm * 64 + i * 16 + hi * 8 + r;
        float vv = acc[i][j][r] + bv;
        if (act) vv = vv > 0.0f ? vv : 0.01f * vv;
        C[(size_t)row * N + col] = vv;
      }
    }
  }
}

// ---------------------------------------------------------------------------
// Weight pre-convert: W[K,N] fp32 row-major -> Wt[N,K] f16 (transposed)
// ---------------------------------------------------------------------------
__global__ void wcvt_t(const float* __restrict__ W, _Float16* __restrict__ Wt,
                       int K, int N)
{
  int idx = blockIdx.x * blockDim.x + threadIdx.x;
  if (idx >= K * N) return;
  int k = idx / N, n = idx - k * N;               // n fastest -> coalesced read
  Wt[(size_t)n * K + k] = (_Float16)W[(size_t)k * N + n];
}

// ---------------------------------------------------------------------------
// LayerNorm over rows of 256: one wave per row.
// ---------------------------------------------------------------------------
__global__ __launch_bounds__(256) void layernorm_k(
    const float* __restrict__ h, float* __restrict__ hn,
    const float* __restrict__ g, const float* __restrict__ b, int NN)
{
  int row  = blockIdx.x * 8 + (threadIdx.x >> 5);
  int lane = threadIdx.x & 31;
  if (row >= NN) return;
  const float* hp = h + (size_t)row * HID;
  float v[8];
  float s = 0.0f;
#pragma unroll
  for (int i = 0; i < 8; ++i) { v[i] = hp[lane + 32 * i]; s += v[i]; }
#pragma unroll
  for (int o = 16; o; o >>= 1) s += __shfl_xor(s, o, 32);
  float mu = s * (1.0f / HID);
  float vs = 0.0f;
#pragma unroll
  for (int i = 0; i < 8; ++i) { float d = v[i] - mu; vs += d * d; }
#pragma unroll
  for (int o = 16; o; o >>= 1) vs += __shfl_xor(vs, o, 32);
  float rs = rsqrtf(vs * (1.0f / HID) + 1e-5f);
  float* op = hn + (size_t)row * HID;
#pragma unroll
  for (int i = 0; i < 8; ++i) {
    int c = lane + 32 * i;
    op[c] = (v[i] - mu) * rs * g[c] + b[c];
  }
}

// ---------------------------------------------------------------------------
// CSR build (fixed-capacity rows) for incoming edges per destination node.
// ---------------------------------------------------------------------------
__global__ void zero_i32(int* p, int n) {
  int i = blockIdx.x * blockDim.x + threadIdx.x;
  if (i < n) p[i] = 0;
}

__global__ void build_csr(const int* __restrict__ ei, int* __restrict__ cnt,
                          int* __restrict__ csr, int E0, int NN)
{
  int e = blockIdx.x * blockDim.x + threadIdx.x;
  if (e >= E0 + NN) return;
  int s, d;
  if (e < E0) { s = ei[e]; d = ei[E0 + e]; }   // [2,E0] flat: row0=src,row1=dst
  else        { s = d = e - E0; }              // self loops
  int slot = atomicAdd(&cnt[d], 1);
  if (slot < CAP) csr[d * CAP + slot] = s;
}

// ---------------------------------------------------------------------------
// GATv2 softmax + aggregation for one head; accumulates into hc.
// One block (256 thr) per destination node; no atomics.
// ---------------------------------------------------------------------------
__global__ __launch_bounds__(256) void gat_agg(
    const float* __restrict__ xl, const float* __restrict__ xr,
    const float* __restrict__ att, const int* __restrict__ csr,
    const int* __restrict__ cnt, float* __restrict__ hc, int headIdx)
{
  __shared__ int   sSrc[CAP];
  __shared__ float sLog[CAP];
  __shared__ float sXr[HID];
  __shared__ float sAtt[HID];

  const int v    = blockIdx.x;
  const int t    = threadIdx.x;
  const int lane = t & 31;
  const int wave = t >> 5;
  int deg = cnt[v]; if (deg > CAP) deg = CAP;

  sXr[t]  = xr[(size_t)v * HID + t];
  sAtt[t] = att[t];
  if (t < deg) sSrc[t] = csr[v * CAP + t];
  __syncthreads();

  for (int e = wave; e < deg; e += 8) {
    const float* xp = xl + (size_t)sSrc[e] * HID;
    float p = 0.0f;
#pragma unroll
    for (int i = 0; i < 8; ++i) {
      int c = lane + 32 * i;
      float tv = xp[c] + sXr[c];
      tv = tv > 0.0f ? tv : 0.2f * tv;
      p += tv * sAtt[c];
    }
#pragma unroll
    for (int o = 16; o; o >>= 1) p += __shfl_xor(p, o, 32);
    if (lane == 0) sLog[e] = p;
  }
  __syncthreads();

  if (wave == 0) {
    float m = -1e30f;
    for (int e = lane; e < deg; e += 32) m = fmaxf(m, sLog[e]);
#pragma unroll
    for (int o = 16; o; o >>= 1) m = fmaxf(m, __shfl_xor(m, o, 32));
    float s = 0.0f;
    for (int e = lane; e < deg; e += 32) {
      float ex = expf(sLog[e] - m);
      sLog[e] = ex; s += ex;
    }
#pragma unroll
    for (int o = 16; o; o >>= 1) s += __shfl_xor(s, o, 32);
    float inv = 1.0f / s;
    for (int e = lane; e < deg; e += 32) sLog[e] *= inv;
  }
  __syncthreads();

  float acc = 0.0f;
  for (int e = 0; e < deg; ++e)
    acc += sLog[e] * xl[(size_t)sSrc[e] * HID + t];

  size_t idx = (size_t)v * HID + t;
  hc[idx] = (headIdx == 0) ? acc : (hc[idx] + acc);
}

// h = lrelu(hc/H + bias) + h
__global__ void residual_k(float* __restrict__ h, const float* __restrict__ hc,
                           const float* __restrict__ bias, float invH, int total)
{
  int i = blockIdx.x * blockDim.x + threadIdx.x;
  if (i >= total) return;
  float v = hc[i] * invH + bias[i & (HID - 1)];
  v = v > 0.0f ? v : 0.01f * v;
  h[i] += v;
}

// precompute gather bases for the edge-feature MLP input
__global__ void edge_bases(const int* __restrict__ pairs, int* __restrict__ b0,
                           int* __restrict__ b1, int NEH, int ROWS)
{
  int r = blockIdx.x * blockDim.x + threadIdx.x;
  if (r >= ROWS) return;
  int g = r / NEH, k = r - g * NEH;
  b0[r] = (g * 30 + pairs[2 * k])     << 8;   // *HID
  b1[r] = (g * 30 + pairs[2 * k + 1]) << 8;
}

// final K=128 -> 1 projection: one wave per row
__global__ __launch_bounds__(256) void mlp_out_k(
    const float* __restrict__ o2, const float* __restrict__ w,
    const float* __restrict__ b, float* __restrict__ out, int ROWS)
{
  int row  = blockIdx.x * 8 + (threadIdx.x >> 5);
  int lane = threadIdx.x & 31;
  if (row >= ROWS) return;
  const float* p = o2 + (size_t)row * 128;
  float acc = 0.0f;
#pragma unroll
  for (int i = 0; i < 4; ++i) acc += p[lane + 32 * i] * w[lane + 32 * i];
#pragma unroll
  for (int o = 16; o; o >>= 1) acc += __shfl_xor(acc, o, 32);
  if (lane == 0) out[row] = acc + b[0];
}

// ---------------------------------------------------------------------------
extern "C" void kernel_launch(void* const* d_in, const int* in_sizes, int n_in,
                              void* d_out, int out_size, void* d_ws, size_t ws_size,
                              hipStream_t stream)
{
  (void)n_in; (void)ws_size;
  const float* x     = (const float*)d_in[0];
  const int*   ei    = (const int*)d_in[1];
  const int*   pairs = (const int*)d_in[2];
  const float* Wp    = (const float*)d_in[3];
  const float* bp    = (const float*)d_in[4];
  const float* Wm0 = (const float*)d_in[29]; const float* bm0 = (const float*)d_in[30];
  const float* Wm1 = (const float*)d_in[31]; const float* bm1 = (const float*)d_in[32];
  const float* Wm2 = (const float*)d_in[33]; const float* bm2 = (const float*)d_in[34];
  const float* Wm3 = (const float*)d_in[35]; const float* bm3 = (const float*)d_in[36];

  const int NN   = in_sizes[0] / 128;   // 30720 nodes
  const int E0   = in_sizes[1] / 2;     // 83968 directed edges
  const int NEH  = in_sizes[2] / 2;     // 41
  const int ROWS = out_size;            // 41984 MLP rows
  const int HEADS[3] = {3, 2, 1};

  // ---- workspace layout ----
  float* h  = (float*)d_ws;                        // NN*256
  float* hn = h  + (size_t)NN * HID;
  float* xl = hn + (size_t)NN * HID;
  float* xr = xl + (size_t)NN * HID;
  float* hc = xr + (size_t)NN * HID;
  int* csr  = (int*)(hc + (size_t)NN * HID);       // NN*CAP
  int* cnt  = csr + (size_t)NN * CAP;              // NN
  int* gb0  = cnt + NN;                            // ROWS
  int* gb1  = gb0 + ROWS;                          // ROWS
  float* o0 = xl;   // ROWS*256 fits in xl+xr (dead after GAT)
  float* o1 = hn;   // ROWS*128
  float* o2 = hc;   // ROWS*128

  // f16 transposed weight copies
  _Float16* wb = (_Float16*)(gb1 + ROWS);
  _Float16* WpT = wb;                       wb += (size_t)128 * 256;
  _Float16* WlT[3]; _Float16* WrT[3];
  for (int l = 0; l < 3; ++l) {
    int Nw = HEADS[l] * HID;
    WlT[l] = wb; wb += (size_t)HID * Nw;
    WrT[l] = wb; wb += (size_t)HID * Nw;
  }
  _Float16* Wm0T = wb; wb += (size_t)512 * 256;
  _Float16* Wm1T = wb; wb += (size_t)256 * 128;
  _Float16* Wm2T = wb; wb += (size_t)128 * 128;

  // ---- graph structure, gather bases, weight conversion ----
  zero_i32<<<(NN + 255) / 256, 256, 0, stream>>>(cnt, NN);
  build_csr<<<(E0 + NN + 255) / 256, 256, 0, stream>>>(ei, cnt, csr, E0, NN);
  edge_bases<<<(ROWS + 255) / 256, 256, 0, stream>>>(pairs, gb0, gb1, NEH, ROWS);

  wcvt_t<<<(128 * 256 + 255) / 256, 256, 0, stream>>>(Wp, WpT, 128, 256);
  for (int l = 0; l < 3; ++l) {
    int Nw = HEADS[l] * HID, tot = HID * Nw;
    const float* Wl = (const float*)d_in[5 + 8 * l + 2];
    const float* Wr = (const float*)d_in[5 + 8 * l + 4];
    wcvt_t<<<(tot + 255) / 256, 256, 0, stream>>>(Wl, WlT[l], HID, Nw);
    wcvt_t<<<(tot + 255) / 256, 256, 0, stream>>>(Wr, WrT[l], HID, Nw);
  }
  wcvt_t<<<(512 * 256 + 255) / 256, 256, 0, stream>>>(Wm0, Wm0T, 512, 256);
  wcvt_t<<<(256 * 128 + 255) / 256, 256, 0, stream>>>(Wm1, Wm1T, 256, 128);
  wcvt_t<<<(128 * 128 + 255) / 256, 256, 0, stream>>>(Wm2, Wm2T, 128, 128);

  // h = x @ Wp + bp
  wmma_gemm_t<false><<<dim3(NN / BM, 2), 256, 0, stream>>>(
      x, WpT, bp, h, NN, HID, 128, 0, nullptr, nullptr);

  for (int l = 0; l < 3; ++l) {
    const int base = 5 + 8 * l;
    const float* lng  = (const float*)d_in[base + 0];
    const float* lnb  = (const float*)d_in[base + 1];
    const float* bl   = (const float*)d_in[base + 3];
    const float* br   = (const float*)d_in[base + 5];
    const float* att  = (const float*)d_in[base + 6];
    const float* bias = (const float*)d_in[base + 7];
    const int H = HEADS[l];

    layernorm_k<<<(NN + 7) / 8, 256, 0, stream>>>(h, hn, lng, lnb, NN);

    for (int hd = 0; hd < H; ++hd) {
      wmma_gemm_t<false><<<dim3(NN / BM, 2), 256, 0, stream>>>(
          hn, WlT[l] + (size_t)hd * HID * HID, bl + hd * HID, xl,
          NN, HID, HID, 0, nullptr, nullptr);
      wmma_gemm_t<false><<<dim3(NN / BM, 2), 256, 0, stream>>>(
          hn, WrT[l] + (size_t)hd * HID * HID, br + hd * HID, xr,
          NN, HID, HID, 0, nullptr, nullptr);
      gat_agg<<<NN, 256, 0, stream>>>(xl, xr, att + hd * HID, csr, cnt, hc, hd);
    }
    residual_k<<<(NN * HID + 255) / 256, 256, 0, stream>>>(
        h, hc, bias, 1.0f / (float)H, NN * HID);
  }

  // edge MLP (first GEMM gathers concat(h[p0], h[p1]) on the fly)
  wmma_gemm_t<true><<<dim3(ROWS / BM, 2), 256, 0, stream>>>(
      h, Wm0T, bm0, o0, ROWS, 256, 512, 1, gb0, gb1);
  wmma_gemm_t<false><<<dim3(ROWS / BM, 1), 256, 0, stream>>>(
      o0, Wm1T, bm1, o1, ROWS, 128, 256, 1, nullptr, nullptr);
  wmma_gemm_t<false><<<dim3(ROWS / BM, 1), 256, 0, stream>>>(
      o1, Wm2T, bm2, o2, ROWS, 128, 128, 1, nullptr, nullptr);
  mlp_out_k<<<(ROWS + 7) / 8, 256, 0, stream>>>(o2, Wm3, bm3, (float*)d_out, ROWS);
}
